// AudioPreviewModel_19756849561940
// MI455X (gfx1250) — compile-verified, matrix operand order checked by reading, not run on hardware
//
#include <hip/hip_runtime.h>
#include <math.h>

// ---------------- model dims (match reference constants) ----------------
#define BB   32
#define TT   1024
#define DD   512
#define HH   1024
#define KK   512
#define CC   400
#define EE   10
#define GH   4096          // 4*H
#define KDC  1536          // D + H (concatenated LSTM input)

typedef __attribute__((ext_vector_type(16))) _Float16 v16h;
typedef __attribute__((ext_vector_type(8)))  _Float16 v8h;
typedef __attribute__((ext_vector_type(8)))  float    v8f;

__device__ __forceinline__ float sigmf(float x) { return 1.0f / (1.0f + expf(-x)); }

// Load one 16x32 f16 WMMA fragment (A or B operand) for this lane.
// p already points at row (tile_row + lane&15), element offset koff.
__device__ __forceinline__ v16h load_frag(const _Float16* __restrict__ p, int kb)
{
    v8h lo = *(const v8h*)(p + kb);       // k = kb+koff+0..7
    v8h hi = *(const v8h*)(p + kb + 16);  // k = kb+koff+16..23
    v16h f;
#pragma unroll
    for (int i = 0; i < 8; ++i) { f[i] = lo[i]; f[i + 8] = hi[i]; }
    return f;
}

// ============================================================
// WMMA GEMM (16x16 per wave): C[M,N] = A[M,Kd]*B[N,Kd]^T + bias
// Used for N % 32 != 0 cases (N = 400).
// ============================================================
__global__ __launch_bounds__(256)
void wmma_gemm_bias(const _Float16* __restrict__ A, const _Float16* __restrict__ B,
                    const float* __restrict__ bias, float* __restrict__ C,
                    int M, int N, int Kd)
{
    int lane = threadIdx.x & 31;
    int wid  = blockIdx.x * (blockDim.x >> 5) + (threadIdx.x >> 5);
    int tilesN = N >> 4;
    int total  = (M >> 4) * tilesN;
    if (wid >= total) return;                 // wave-uniform exit
    int tm = wid / tilesN;
    int tn = wid - tm * tilesN;

    int r15  = lane & 15;
    int koff = (lane >= 16) ? 8 : 0;
    const _Float16* pa = A + (size_t)(tm * 16 + r15) * Kd + koff;
    const _Float16* pb = B + (size_t)(tn * 16 + r15) * Kd + koff;

    v8f acc = {};
    for (int kb = 0; kb < Kd; kb += 32) {
        v16h af = load_frag(pa, kb);
        v16h bf = load_frag(pb, kb);
        acc = __builtin_amdgcn_wmma_f32_16x16x32_f16(
                  false, af, false, bf, (short)0, acc, false, false);
    }

    int n  = tn * 16 + r15;
    float bv = bias ? bias[n] : 0.0f;
    int m0 = tm * 16 + ((lane >= 16) ? 8 : 0);
#pragma unroll
    for (int i = 0; i < 8; ++i)
        C[(size_t)(m0 + i) * N + n] = acc[i] + bv;
}

// ============================================================
// WMMA GEMM, 2x2 register-blocked (32x32 per wave).
// Each A/B fragment feeds two WMMAs: halves L2 fragment traffic.
// Requires M % 32 == 0 and N % 32 == 0.
// ============================================================
__global__ __launch_bounds__(256)
void wmma_gemm_bias2(const _Float16* __restrict__ A, const _Float16* __restrict__ B,
                     const float* __restrict__ bias, float* __restrict__ C,
                     int M, int N, int Kd)
{
    int lane = threadIdx.x & 31;
    int wid  = blockIdx.x * (blockDim.x >> 5) + (threadIdx.x >> 5);
    int tilesN = N >> 5;
    int total  = (M >> 5) * tilesN;
    if (wid >= total) return;                 // wave-uniform exit
    int tm = wid / tilesN;
    int tn = wid - tm * tilesN;

    int r15  = lane & 15;
    int koff = (lane >= 16) ? 8 : 0;
    const _Float16* pa0 = A + (size_t)(tm * 32 + r15)      * Kd + koff;
    const _Float16* pa1 = A + (size_t)(tm * 32 + 16 + r15) * Kd + koff;
    const _Float16* pb0 = B + (size_t)(tn * 32 + r15)      * Kd + koff;
    const _Float16* pb1 = B + (size_t)(tn * 32 + 16 + r15) * Kd + koff;

    v8f acc00 = {}, acc01 = {}, acc10 = {}, acc11 = {};
    for (int kb = 0; kb < Kd; kb += 32) {
        v16h a0 = load_frag(pa0, kb);
        v16h a1 = load_frag(pa1, kb);
        v16h b0 = load_frag(pb0, kb);
        v16h b1 = load_frag(pb1, kb);
        acc00 = __builtin_amdgcn_wmma_f32_16x16x32_f16(false, a0, false, b0, (short)0, acc00, false, false);
        acc01 = __builtin_amdgcn_wmma_f32_16x16x32_f16(false, a0, false, b1, (short)0, acc01, false, false);
        acc10 = __builtin_amdgcn_wmma_f32_16x16x32_f16(false, a1, false, b0, (short)0, acc10, false, false);
        acc11 = __builtin_amdgcn_wmma_f32_16x16x32_f16(false, a1, false, b1, (short)0, acc11, false, false);
    }

    int n0 = tn * 32 + r15;
    int n1 = n0 + 16;
    float bv0 = bias ? bias[n0] : 0.0f;
    float bv1 = bias ? bias[n1] : 0.0f;
    int m0 = tm * 32 + ((lane >= 16) ? 8 : 0);
#pragma unroll
    for (int i = 0; i < 8; ++i) {
        C[(size_t)(m0 + i) * N + n0]      = acc00[i] + bv0;
        C[(size_t)(m0 + i) * N + n1]      = acc01[i] + bv1;
        C[(size_t)(m0 + 16 + i) * N + n0] = acc10[i] + bv0;
        C[(size_t)(m0 + 16 + i) * N + n1] = acc11[i] + bv1;
    }
}

// ============================================================
// Weight-norm prep: W = g * v / ||v||_row, stored f16 into
// Wcomb rows [W_ih | W_hh] (GH x KDC); bias_comb = b_ih+b_hh
// ============================================================
__global__ __launch_bounds__(256)
void lstm_weight_prep(const float* __restrict__ v_ih, const float* __restrict__ g_ih,
                      const float* __restrict__ v_hh, const float* __restrict__ g_hh,
                      const float* __restrict__ b_ih, const float* __restrict__ b_hh,
                      _Float16* __restrict__ Wcomb, float* __restrict__ bias_comb)
{
    int r = blockIdx.x;          // 0..GH-1
    int t = threadIdx.x;         // 0..255
    __shared__ float red[256];

    float s = 0.f;
    for (int j = t; j < DD; j += 256) { float v = v_ih[(size_t)r * DD + j]; s += v * v; }
    red[t] = s; __syncthreads();
    for (int o = 128; o > 0; o >>= 1) { if (t < o) red[t] += red[t + o]; __syncthreads(); }
    float sc_ih = g_ih[r] / sqrtf(red[0]);
    __syncthreads();

    s = 0.f;
    for (int j = t; j < HH; j += 256) { float v = v_hh[(size_t)r * HH + j]; s += v * v; }
    red[t] = s; __syncthreads();
    for (int o = 128; o > 0; o >>= 1) { if (t < o) red[t] += red[t + o]; __syncthreads(); }
    float sc_hh = g_hh[r] / sqrtf(red[0]);

    for (int j = t; j < DD; j += 256)
        Wcomb[(size_t)r * KDC + j] = (_Float16)(v_ih[(size_t)r * DD + j] * sc_ih);
    for (int j = t; j < HH; j += 256)
        Wcomb[(size_t)r * KDC + DD + j] = (_Float16)(v_hh[(size_t)r * HH + j] * sc_hh);
    if (t == 0) bias_comb[r] = b_ih[r] + b_hh[r];
}

// ---------------- small utility kernels ----------------
__global__ void f32_to_f16_kernel(_Float16* __restrict__ dst, const float* __restrict__ src, size_t n)
{
    size_t i = (size_t)blockIdx.x * blockDim.x + threadIdx.x;
    if (i < n) dst[i] = (_Float16)src[i];
}
__global__ void zero_f32_kernel(float* p, size_t n)
{
    size_t i = (size_t)blockIdx.x * blockDim.x + threadIdx.x;
    if (i < n) p[i] = 0.0f;
}
__global__ void zero_f16_kernel(_Float16* p, size_t n)
{
    size_t i = (size_t)blockIdx.x * blockDim.x + threadIdx.x;
    if (i < n) p[i] = (_Float16)0.0f;
}
__global__ void copy_f32_kernel(float* dst, const float* src, size_t n)
{
    size_t i = (size_t)blockIdx.x * blockDim.x + threadIdx.x;
    if (i < n) dst[i] = src[i];
}

// idx0 = argmax_t(start_rand + masks) (first max); masksum = sum_t masks
__global__ __launch_bounds__(256)
void init_select_kernel(const float* __restrict__ start_rand, const float* __restrict__ masks,
                        int* __restrict__ idxbuf, float* __restrict__ masksum)
{
    int b = blockIdx.x, t0 = threadIdx.x;
    __shared__ float sv[256]; __shared__ int si[256]; __shared__ float ss[256];
    float bv = -INFINITY; int bi = 0; float ms = 0.f;
    for (int t = t0; t < TT; t += 256) {
        float mv = masks[b * TT + t];
        float v  = start_rand[b * TT + t] + mv;
        ms += mv;
        if (v > bv) { bv = v; bi = t; }
    }
    sv[t0] = bv; si[t0] = bi; ss[t0] = ms; __syncthreads();
    for (int o = 128; o > 0; o >>= 1) {
        if (t0 < o) {
            if (sv[t0 + o] > sv[t0] || (sv[t0 + o] == sv[t0] && si[t0 + o] < si[t0])) {
                sv[t0] = sv[t0 + o]; si[t0] = si[t0 + o];
            }
            ss[t0] += ss[t0 + o];
        }
        __syncthreads();
    }
    if (t0 == 0) { idxbuf[b] = si[0]; masksum[b] = ss[0]; }
}

// feat0[b,d] = (sum_t features[b,t,d]) / masksum[b]  -> f16 into Acomb[:, 0:D]
__global__ void feat_mean0_kernel(const float* __restrict__ features,
                                  const float* __restrict__ masksum,
                                  _Float16* __restrict__ Acomb)
{
    int i = blockIdx.x * blockDim.x + threadIdx.x;   // B*D
    if (i >= BB * DD) return;
    int b = i >> 9, d = i & (DD - 1);
    const float* f = features + (size_t)b * TT * DD + d;
    float s = 0.f;
    for (int t = 0; t < TT; ++t) s += f[(size_t)t * DD];
    Acomb[b * KDC + d] = (_Float16)(s / masksum[b]);
}

__global__ void mask_update_kernel(float* masks, const int* idx)
{
    int b = threadIdx.x;
    if (b < BB) masks[b * TT + idx[b]] = 0.0f;
}

// gates(32x4096) -> c2, h2; h2 written f16 into h2f16 and Acomb[:, D:D+H]
__global__ void lstm_pointwise_kernel(const float* __restrict__ gates, float* __restrict__ c,
                                      _Float16* __restrict__ h2f16, _Float16* __restrict__ Acomb)
{
    int i = blockIdx.x * blockDim.x + threadIdx.x;   // B*H
    if (i >= BB * HH) return;
    int b = i >> 10, j = i & (HH - 1);
    const float* g = gates + (size_t)b * GH;
    float gi = g[j], gf = g[HH + j], gg = g[2 * HH + j], go = g[3 * HH + j];
    float c2 = sigmf(gf) * c[i] + sigmf(gi) * tanhf(gg);
    float h2 = sigmf(go) * tanhf(c2);
    c[i] = c2;
    _Float16 hh = (_Float16)h2;
    h2f16[i] = hh;
    Acomb[b * KDC + DD + j] = hh;
}

__global__ void bn_relu_kernel(const float* __restrict__ z,
                               const float* __restrict__ gamma, const float* __restrict__ beta,
                               const float* __restrict__ mean,  const float* __restrict__ var,
                               _Float16* __restrict__ zf16)
{
    int i = blockIdx.x * blockDim.x + threadIdx.x;   // B*H
    if (i >= BB * HH) return;
    int j = i & (HH - 1);
    float zz = (z[i] - mean[j]) * rsqrtf(var[j] + 1e-5f) * gamma[j] + beta[j];
    zf16[i] = (_Float16)fmaxf(zz, 0.0f);
}

// scores[b,t] = scale * dot(q[b,:], FK[b*T+t,:]) * m + (1-m)*(-1e35); one wave per (b,t)
__global__ __launch_bounds__(256)
void scores_kernel(const float* __restrict__ q, const float* __restrict__ FK,
                   const float* __restrict__ masks, float* __restrict__ scores, float scale)
{
    int gid  = blockIdx.x * blockDim.x + threadIdx.x;
    int wave = gid >> 5, lane = gid & 31;
    if (wave >= BB * TT) return;
    int b = wave >> 10;
    const float* qr = q  + (size_t)b * KK;
    const float* fk = FK + (size_t)wave * KK;
    float s = 0.f;
    for (int i = lane; i < KK; i += 32) s += qr[i] * fk[i];
    for (int o = 16; o > 0; o >>= 1) s += __shfl_xor(s, o);
    if (lane == 0) {
        float m = masks[wave];
        scores[wave] = s * scale * m + (1.0f - m) * (-1e35f);
    }
}

// softmax over t, argmax -> idx, and gtsum[b,:] += gt[b, idx, :]
__global__ __launch_bounds__(256)
void softmax_select_kernel(const float* __restrict__ scores, float* __restrict__ probs,
                           int* __restrict__ idxbuf, const float* __restrict__ gt,
                           float* __restrict__ gtsum)
{
    int b = blockIdx.x, t0 = threadIdx.x;
    __shared__ float sv[256]; __shared__ int si[256];
    const float* sc = scores + (size_t)b * TT;

    float bv = -INFINITY; int bi = 0;
    for (int t = t0; t < TT; t += 256) { float v = sc[t]; if (v > bv) { bv = v; bi = t; } }
    sv[t0] = bv; si[t0] = bi; __syncthreads();
    for (int o = 128; o > 0; o >>= 1) {
        if (t0 < o) {
            if (sv[t0 + o] > sv[t0] || (sv[t0 + o] == sv[t0] && si[t0 + o] < si[t0])) {
                sv[t0] = sv[t0 + o]; si[t0] = si[t0 + o];
            }
        }
        __syncthreads();
    }
    float mx = sv[0]; int am = si[0];
    __syncthreads();

    float ls = 0.f;
    for (int t = t0; t < TT; t += 256) {
        float e = expf(sc[t] - mx);
        probs[(size_t)b * TT + t] = e;
        ls += e;
    }
    sv[t0] = ls; __syncthreads();
    for (int o = 128; o > 0; o >>= 1) { if (t0 < o) sv[t0] += sv[t0 + o]; __syncthreads(); }
    float inv = 1.0f / sv[0];
    for (int t = t0; t < TT; t += 256) probs[(size_t)b * TT + t] *= inv;

    if (t0 == 0) idxbuf[b] = am;
    const float* g = gt + ((size_t)b * TT + am) * CC;
    for (int cix = t0; cix < CC; cix += 256)
        gtsum[b * CC + cix] += g[cix];
}

// feat2[b,d] = sum_t probs[b,t]*features[b,t,d]; write f16 into Acomb[:,0:D], accumulate featsum
__global__ void feat2_kernel(const float* __restrict__ features, const float* __restrict__ probs,
                             _Float16* __restrict__ Acomb, float* __restrict__ featsum)
{
    int i = blockIdx.x * blockDim.x + threadIdx.x;   // B*D
    if (i >= BB * DD) return;
    int b = i >> 9, d = i & (DD - 1);
    const float* f = features + (size_t)b * TT * DD + d;
    const float* p = probs + (size_t)b * TT;
    float s = 0.f;
    for (int t = 0; t < TT; ++t) s += p[t] * f[(size_t)t * DD];
    Acomb[b * KDC + d] = (_Float16)s;
    featsum[i] += s;
}

__global__ void feat_final_kernel(const float* __restrict__ featsum, _Float16* __restrict__ fmF16)
{
    int i = blockIdx.x * blockDim.x + threadIdx.x;   // B*D
    if (i >= BB * DD) return;
    fmF16[i] = (_Float16)(featsum[i] * (1.0f / (float)EE));
}
__global__ void gt_final_kernel(const float* __restrict__ gtsum, float* __restrict__ out)
{
    int i = blockIdx.x * blockDim.x + threadIdx.x;   // B*C
    if (i >= BB * CC) return;
    out[i] = gtsum[i] * (1.0f / (float)EE);
}

// ============================================================
// host-side launch
// ============================================================
static inline int gemm_blocks16(int M, int N) { return (((M >> 4) * (N >> 4)) + 7) / 8; }
static inline int gemm_blocks32(int M, int N) { return (((M >> 5) * (N >> 5)) + 7) / 8; }

extern "C" void kernel_launch(void* const* d_in, const int* in_sizes, int n_in,
                              void* d_out, int out_size, void* d_ws, size_t ws_size,
                              hipStream_t stream)
{
    (void)in_sizes; (void)n_in; (void)out_size; (void)ws_size;
    const float* features   = (const float*)d_in[0];
    const float* gt         = (const float*)d_in[1];
    const float* masks_in   = (const float*)d_in[2];
    const float* start_rand = (const float*)d_in[3];
    const float* v_ih = (const float*)d_in[4];
    const float* g_ih = (const float*)d_in[5];
    const float* b_ih = (const float*)d_in[6];
    const float* v_hh = (const float*)d_in[7];
    const float* g_hh = (const float*)d_in[8];
    const float* b_hh = (const float*)d_in[9];
    const float* W1 = (const float*)d_in[10];
    const float* b1 = (const float*)d_in[11];
    const float* bn_gamma = (const float*)d_in[12];
    const float* bn_beta  = (const float*)d_in[13];
    const float* bn_mean  = (const float*)d_in[14];
    const float* bn_var   = (const float*)d_in[15];
    const float* W2 = (const float*)d_in[16];
    const float* b2 = (const float*)d_in[17];
    const float* Wp = (const float*)d_in[18];
    const float* bp = (const float*)d_in[19];
    const float* Wk = (const float*)d_in[20];
    const float* bk = (const float*)d_in[21];
    const float* Wc = (const float*)d_in[22];
    const float* bc = (const float*)d_in[23];
    float* out = (float*)d_out;
    float* out_pred = out;                       // (B,C)
    float* out_gt   = out + BB * CC;             // (B,C)
    float* out_sp   = out + 2 * BB * CC;         // (E,B,C)

    // ---- workspace carve-up ----
    char* w = (char*)d_ws;
    auto carve = [&](size_t bytes) -> void* {
        void* p = (void*)w;
        w += (bytes + 255) & ~(size_t)255;
        return p;
    };
    _Float16* featF16 = (_Float16*)carve((size_t)BB * TT * DD * 2);
    float*    FK      = (float*)   carve((size_t)BB * TT * KK * 4);
    _Float16* Wcomb   = (_Float16*)carve((size_t)GH * KDC * 2);
    float*    biascmb = (float*)   carve((size_t)GH * 4);
    _Float16* WkF16   = (_Float16*)carve((size_t)KK * DD * 2);
    _Float16* W1F16   = (_Float16*)carve((size_t)HH * HH * 2);
    _Float16* W2F16   = (_Float16*)carve((size_t)KK * HH * 2);
    _Float16* WpF16   = (_Float16*)carve((size_t)CC * HH * 2);
    _Float16* WcF16   = (_Float16*)carve((size_t)CC * DD * 2);
    _Float16* Acomb   = (_Float16*)carve((size_t)BB * KDC * 2);
    float*    gates   = (float*)   carve((size_t)BB * GH * 4);
    float*    cstate  = (float*)   carve((size_t)BB * HH * 4);
    _Float16* h2F16   = (_Float16*)carve((size_t)BB * HH * 2);
    float*    zbuf    = (float*)   carve((size_t)BB * HH * 4);
    _Float16* zF16    = (_Float16*)carve((size_t)BB * HH * 2);
    float*    qbuf    = (float*)   carve((size_t)BB * KK * 4);
    float*    scores  = (float*)   carve((size_t)BB * TT * 4);
    float*    probs   = (float*)   carve((size_t)BB * TT * 4);
    float*    masks   = (float*)   carve((size_t)BB * TT * 4);
    float*    masksum = (float*)   carve((size_t)BB * 4);
    int*      idxbuf  = (int*)     carve((size_t)BB * 4);
    float*    featsum = (float*)   carve((size_t)BB * DD * 4);
    float*    gtsum   = (float*)   carve((size_t)BB * CC * 4);
    _Float16* fmF16   = (_Float16*)carve((size_t)BB * DD * 2);

    const float scale = 1.0f / sqrtf((float)KK);
    auto nb = [](size_t n) { return (int)((n + 255) / 256); };

    // ---- one-time-per-call prep ----
    lstm_weight_prep<<<GH, 256, 0, stream>>>(v_ih, g_ih, v_hh, g_hh, b_ih, b_hh, Wcomb, biascmb);
    f32_to_f16_kernel<<<nb((size_t)KK * DD), 256, 0, stream>>>(WkF16, Wk, (size_t)KK * DD);
    f32_to_f16_kernel<<<nb((size_t)HH * HH), 256, 0, stream>>>(W1F16, W1, (size_t)HH * HH);
    f32_to_f16_kernel<<<nb((size_t)KK * HH), 256, 0, stream>>>(W2F16, W2, (size_t)KK * HH);
    f32_to_f16_kernel<<<nb((size_t)CC * HH), 256, 0, stream>>>(WpF16, Wp, (size_t)CC * HH);
    f32_to_f16_kernel<<<nb((size_t)CC * DD), 256, 0, stream>>>(WcF16, Wc, (size_t)CC * DD);
    f32_to_f16_kernel<<<nb((size_t)BB * TT * DD), 256, 0, stream>>>(featF16, features, (size_t)BB * TT * DD);

    zero_f16_kernel<<<nb((size_t)BB * KDC), 256, 0, stream>>>(Acomb, (size_t)BB * KDC);
    zero_f32_kernel<<<nb((size_t)BB * HH), 256, 0, stream>>>(cstate, (size_t)BB * HH);
    zero_f32_kernel<<<nb((size_t)BB * DD), 256, 0, stream>>>(featsum, (size_t)BB * DD);
    zero_f32_kernel<<<nb((size_t)BB * CC), 256, 0, stream>>>(gtsum, (size_t)BB * CC);
    copy_f32_kernel<<<nb((size_t)BB * TT), 256, 0, stream>>>(masks, masks_in, (size_t)BB * TT);

    init_select_kernel<<<BB, 256, 0, stream>>>(start_rand, masks_in, idxbuf, masksum);
    feat_mean0_kernel<<<nb(BB * DD), 256, 0, stream>>>(features, masksum, Acomb);

    // feature_keys: FK[(b*T+t), k] = features[b,t,:] . Wk[k,:] + bk[k]  (2x2 blocked)
    wmma_gemm_bias2<<<gemm_blocks32(BB * TT, KK), 256, 0, stream>>>(
        featF16, WkF16, bk, FK, BB * TT, KK, DD);

    // ---- episode loop (E fixed = 10 per reference) ----
    for (int s = 0; s < EE; ++s) {
        mask_update_kernel<<<1, 32, 0, stream>>>(masks, idxbuf);

        // gates = [feat|h] @ [W_ih|W_hh]^T + (b_ih + b_hh)
        wmma_gemm_bias2<<<gemm_blocks32(BB, GH), 256, 0, stream>>>(
            Acomb, Wcomb, biascmb, gates, BB, GH, KDC);
        lstm_pointwise_kernel<<<nb(BB * HH), 256, 0, stream>>>(gates, cstate, h2F16, Acomb);

        // MLP: z = h2@W1^T + b1 ; BN ; ReLU ; q = .@W2^T + b2
        wmma_gemm_bias2<<<gemm_blocks32(BB, HH), 256, 0, stream>>>(
            h2F16, W1F16, b1, zbuf, BB, HH, HH);
        bn_relu_kernel<<<nb(BB * HH), 256, 0, stream>>>(zbuf, bn_gamma, bn_beta, bn_mean, bn_var, zF16);
        wmma_gemm_bias2<<<gemm_blocks32(BB, KK), 256, 0, stream>>>(
            zF16, W2F16, b2, qbuf, BB, KK, HH);

        // step_pred = h2@Wp^T + bp  (written straight into d_out; N=400 -> 16x16 kernel)
        wmma_gemm_bias<<<gemm_blocks16(BB, CC), 256, 0, stream>>>(
            h2F16, WpF16, bp, out_sp + (size_t)s * BB * CC, BB, CC, HH);

        // attention scores / softmax / select / pooled feature
        scores_kernel<<<(BB * TT * 32) / 256, 256, 0, stream>>>(qbuf, FK, masks, scores, scale);
        softmax_select_kernel<<<BB, 256, 0, stream>>>(scores, probs, idxbuf, gt, gtsum);
        feat2_kernel<<<nb(BB * DD), 256, 0, stream>>>(features, probs, Acomb, featsum);
    }

    // ---- finals ----
    feat_final_kernel<<<nb(BB * DD), 256, 0, stream>>>(featsum, fmF16);
    gt_final_kernel<<<nb(BB * CC), 256, 0, stream>>>(gtsum, out_gt);
    wmma_gemm_bias<<<gemm_blocks16(BB, CC), 256, 0, stream>>>(
        fmF16, WcF16, bc, out_pred, BB, CC, DD);
}
